// Model_83786222010693
// MI455X (gfx1250) — matrix operand order repaired and checked by reading, not compiled, not run on hardware
//
#include <hip/hip_runtime.h>
#include <hip/hip_bf16.h>

// ---------------------------------------------------------------------------
// GAT (2-layer, DGL GATConv style) for MI455X / gfx1250.
// Dense projections: v_wmma_f32_16x16x32_bf16, one wave per 16-row strip,
// NT N-tiles per wave (A fragment amortized), A fragment loads are aligned
// global_load_b128 from pre-padded bf16 buffers (no guards in the K-loop).
// Edge softmax: 3 streaming edge passes with f32 atomics; segment-max via
// order-preserving uint encoding + atomicMax.
// ---------------------------------------------------------------------------

typedef __attribute__((ext_vector_type(16))) __bf16 v16bf;
typedef __attribute__((ext_vector_type(8)))  __bf16 v8bf;
typedef __attribute__((ext_vector_type(8)))  float  v8f;

#define N_NODES 50000
#define N_EDGES 1600000
#define IN0     1433
#define K0P     1440      // IN0 padded to multiple of 32
#define F0      64        // H0*D0
#define H0      8
#define D0      8
#define IN1     64
#define H1      1
#define D1      7
#define NP1     16        // D1 padded to 16 for WMMA N-tile
#define NEG_SLOPE 0.2f
#define ENC_NEG_INF 0x007FFFFFu   // enc(-inf)

// ---- order-preserving float<->uint encoding (monotone) --------------------
__device__ __forceinline__ unsigned enc_f32(float f) {
  unsigned u = __float_as_uint(f);
  return (u & 0x80000000u) ? ~u : (u | 0x80000000u);
}

// ---- fill words -----------------------------------------------------------
__global__ void fill_u32(unsigned* __restrict__ p, unsigned v, size_t n) {
  size_t i = (size_t)blockIdx.x * blockDim.x + threadIdx.x;
  if (i < n) p[i] = v;
}

// ---- f32 -> bf16 with zero padding ----------------------------------------
__global__ void cvt_pad_bf16(const float* __restrict__ s, int rows, int cols,
                             int pcols, __bf16* __restrict__ d, size_t total) {
  size_t i = (size_t)blockIdx.x * blockDim.x + threadIdx.x;
  if (i >= total) return;
  int r = (int)(i / pcols), c = (int)(i % pcols);
  float v = (c < cols && r < rows) ? s[(size_t)r * cols + c] : 0.0f;
  d[i] = (__bf16)v;
}

// ---- WMMA GEMM: C[M,ldc] = A[M,Kp](bf16, padded) x B[Kp,ldb](bf16) --------
// grid.x = M/16 strips, block = 32 (one wave). Each wave computes NT 16x16
// D-tiles (N = NT*16 <= ldb), loading the A fragment once per K-step.
// A fragment (16-bit 16x32 ISA layout): lane m=L%16, hi=L>=16;
//   elems 0..7  = A[m][kb + hi*8 + 0..7]      (VGPRs 0-3)
//   elems 8..15 = A[m][kb + 16 + hi*8 + 0..7] (VGPRs 4-7)
// B fragment: col n=t*16+L%16, elems 0..7 = B[kb+hi*8+0..7][n], 8..15 = B[kb+16+hi*8+0..7][n].
template <int NT>
__global__ void gemm_wmma_bf16(const __bf16* __restrict__ A, int lda, int Kp,
                               const __bf16* __restrict__ B, int ldb,
                               float* __restrict__ C, int ldc) {
  const int lane  = threadIdx.x & 31;
  const int mtile = blockIdx.x;
  const int m  = lane & 15;
  const int hi = (lane >> 4) & 1;
  const __bf16* __restrict__ Arow = A + (size_t)(mtile * 16 + m) * lda + hi * 8;

  v8f acc[NT];
#pragma unroll
  for (int t = 0; t < NT; ++t) acc[t] = (v8f){};

  for (int kb = 0; kb < Kp; kb += 32) {
    union { v16bf v; v8bf h[2]; } a;
    a.h[0] = *(const v8bf*)(Arow + kb);        // 16B aligned
    a.h[1] = *(const v8bf*)(Arow + kb + 16);
    const __bf16* __restrict__ Brow = B + (size_t)(kb + hi * 8) * ldb + (lane & 15);
#pragma unroll
    for (int t = 0; t < NT; ++t) {
      union { v16bf v; __bf16 e[16]; } b;
#pragma unroll
      for (int i = 0; i < 8; ++i) {
        b.e[i]     = Brow[(size_t)i * ldb + t * 16];
        b.e[8 + i] = Brow[(size_t)(16 + i) * ldb + t * 16];
      }
      acc[t] = __builtin_amdgcn_wmma_f32_16x16x32_bf16(
          /*neg_a=*/false, a.v, /*neg_b=*/false, b.v,
          /*c_mod=*/(short)0, acc[t], /*reuse_a=*/false, /*reuse_b=*/false);
    }
  }

#pragma unroll
  for (int t = 0; t < NT; ++t) {
    union { v8f v; float e[8]; } c;
    c.v = acc[t];
    float* __restrict__ Crow = C + (size_t)(mtile * 16 + hi * 8) * ldc
                                 + t * 16 + (lane & 15);
#pragma unroll
    for (int r = 0; r < 8; ++r) Crow[(size_t)r * ldc] = c.e[r];
  }
}

// ---- per-node attention logits: el/er[n*H+h] = <h[n,h,:], a[h,:]> ---------
__global__ void node_el_er(const float* __restrict__ hlin, int S,
                           const float* __restrict__ al,
                           const float* __restrict__ ar,
                           int NH, int H, int D,
                           float* __restrict__ el, float* __restrict__ er) {
  int i = blockIdx.x * blockDim.x + threadIdx.x;
  if (i >= NH) return;
  int n = i / H, h = i % H;
  const float* __restrict__ hp  = hlin + (size_t)n * S + h * D;
  const float* __restrict__ alp = al + h * D;
  const float* __restrict__ arp = ar + h * D;
  float sl = 0.f, sr = 0.f;
  for (int d = 0; d < D; ++d) { float x = hp[d]; sl += x * alp[d]; sr += x * arp[d]; }
  el[i] = sl; er[i] = sr;
}

// ---- edge pass 1: e = leaky_relu(el[src]+er[dst]); segment max ------------
__global__ void edge_lrelu_max(const int* __restrict__ src, const int* __restrict__ dst,
                               const float* __restrict__ el, const float* __restrict__ er,
                               int E, int H, float* __restrict__ ebuf,
                               unsigned* __restrict__ emax_enc) {
  int e = blockIdx.x * blockDim.x + threadIdx.x;
  if (e >= E) return;
  int s = src[e], d = dst[e];
  for (int h = 0; h < H; ++h) {
    float x = el[(size_t)s * H + h] + er[(size_t)d * H + h];
    float f = (x > 0.f) ? x : NEG_SLOPE * x;
    ebuf[(size_t)e * H + h] = f;
    atomicMax(&emax_enc[(size_t)d * H + h], enc_f32(f));
  }
}

// ---- decode enc(max) -> float, empty/non-finite segments -> 0 -------------
__global__ void decode_emax(unsigned* __restrict__ p, size_t n) {
  size_t i = (size_t)blockIdx.x * blockDim.x + threadIdx.x;
  if (i >= n) return;
  unsigned u = p[i];
  unsigned bits = (u & 0x80000000u) ? (u ^ 0x80000000u) : ~u;
  float f = __uint_as_float(bits);
  if ((bits & 0x7F800000u) == 0x7F800000u) f = 0.0f;   // !isfinite -> 0
  p[i] = __float_as_uint(f);
}

// ---- edge pass 2: ee = exp(e - emax[dst]); segment sum --------------------
__global__ void edge_exp_sum(const int* __restrict__ dst,
                             const float* __restrict__ emax,
                             int E, int H, float* __restrict__ ebuf,
                             float* __restrict__ den) {
  int e = blockIdx.x * blockDim.x + threadIdx.x;
  if (e >= E) return;
  int d = dst[e];
  for (int h = 0; h < H; ++h) {
    float ee = __expf(ebuf[(size_t)e * H + h] - emax[(size_t)d * H + h]);
    ebuf[(size_t)e * H + h] = ee;
    atomicAdd(&den[(size_t)d * H + h], ee);
  }
}

// ---- edge pass 3: out[dst,h,:] += (ee/den[dst,h]) * h[src,h,:] ------------
// one thread per (edge, head)
__global__ void edge_aggregate(const int* __restrict__ src, const int* __restrict__ dst,
                               const float* __restrict__ ebuf,
                               const float* __restrict__ den,
                               const float* __restrict__ hlin, int S,
                               int EH, int H, int D,
                               float* __restrict__ out, int outS) {
  int i = blockIdx.x * blockDim.x + threadIdx.x;
  if (i >= EH) return;
  int e = i / H, h = i % H;
  int s = src[e], d = dst[e];
  const float* __restrict__ hs = hlin + (size_t)s * S + h * D;
  __builtin_prefetch(hs, 0, 0);            // global_prefetch_b8 on gfx1250
  float alpha = ebuf[i] / den[(size_t)d * H + h];
  float* __restrict__ od = out + (size_t)d * outS + h * D;
  for (int dd = 0; dd < D; ++dd) atomicAdd(&od[dd], alpha * hs[dd]);
}

// ---- ELU, f32 accumulators -> bf16 (feeds layer-1 GEMM A operand) ---------
__global__ void elu_to_bf16(const float* __restrict__ in, __bf16* __restrict__ o,
                            size_t n) {
  size_t i = (size_t)blockIdx.x * blockDim.x + threadIdx.x;
  if (i >= n) return;
  float x = in[i];
  o[i] = (__bf16)((x > 0.f) ? x : expm1f(x));
}

// ---------------------------------------------------------------------------
static inline unsigned cdiv(size_t a, unsigned b) { return (unsigned)((a + b - 1) / b); }

extern "C" void kernel_launch(void* const* d_in, const int* in_sizes, int n_in,
                              void* d_out, int out_size, void* d_ws, size_t ws_size,
                              hipStream_t stream) {
  const float* feat = (const float*)d_in[0];
  const int*   src  = (const int*)  d_in[1];
  const int*   dst  = (const int*)  d_in[2];
  const float* W0   = (const float*)d_in[3];
  const float* al0  = (const float*)d_in[4];
  const float* ar0  = (const float*)d_in[5];
  const float* W1   = (const float*)d_in[6];
  const float* al1  = (const float*)d_in[7];
  const float* ar1  = (const float*)d_in[8];
  float* out = (float*)d_out;

  // ---- workspace carve-out (256B aligned) ----
  char* ws = (char*)d_ws;
  size_t off = 0;
  auto alloc = [&](size_t bytes) -> void* {
    void* p = ws + off;
    off = (off + bytes + 255) & ~(size_t)255;
    return p;
  };
  __bf16*   featb = (__bf16*)  alloc((size_t)N_NODES * K0P * 2);   // [N,1440] bf16
  __bf16*   Wb0   = (__bf16*)  alloc((size_t)K0P * F0 * 2);        // [1440,64] bf16
  float*    hlin0 = (float*)   alloc((size_t)N_NODES * F0 * 4);    // [N,64]
  float*    el0   = (float*)   alloc((size_t)N_NODES * H0 * 4);
  float*    er0   = (float*)   alloc((size_t)N_NODES * H0 * 4);
  unsigned* emax0 = (unsigned*)alloc((size_t)N_NODES * H0 * 4);    // enc -> f32
  float*    den0  = (float*)   alloc((size_t)N_NODES * H0 * 4);
  float*    ebuf0 = (float*)   alloc((size_t)N_EDGES * H0 * 4);    // [E,8]
  float*    hout0 = (float*)   alloc((size_t)N_NODES * F0 * 4);    // [N,64] accum
  __bf16*   h1b   = (__bf16*)  alloc((size_t)N_NODES * IN1 * 2);   // [N,64] bf16 (ELU'd)
  __bf16*   Wb1   = (__bf16*)  alloc((size_t)IN1 * NP1 * 2);       // [64,16] bf16
  float*    hlin1 = (float*)   alloc((size_t)N_NODES * NP1 * 4);   // [N,16]
  float*    el1   = (float*)   alloc((size_t)N_NODES * 4);
  float*    er1   = (float*)   alloc((size_t)N_NODES * 4);
  unsigned* emax1 = (unsigned*)alloc((size_t)N_NODES * 4);
  float*    den1  = (float*)   alloc((size_t)N_NODES * 4);
  float*    ebuf1 = (float*)   alloc((size_t)N_EDGES * 4);
  (void)ws_size; (void)in_sizes; (void)n_in;

  const unsigned B = 256;

  // ==================== Layer 0 ====================
  // inputs -> bf16 (zero-padded K to 1440)
  cvt_pad_bf16<<<cdiv((size_t)N_NODES * K0P, B), B, 0, stream>>>(
      feat, N_NODES, IN0, K0P, featb, (size_t)N_NODES * K0P);
  cvt_pad_bf16<<<cdiv((size_t)K0P * F0, B), B, 0, stream>>>(
      W0, IN0, F0, F0, Wb0, (size_t)K0P * F0);   // pcols==cols; row pad via rows<K0P? no:
  // NOTE: W0 pad handled below via row guard: rows=IN0 < K0P rows in buffer
  // (cvt_pad_bf16 zero-fills r>=rows)
  // hlin0 = feat @ W0   (WMMA, 3125 strips x 4 N-tiles per wave)
  gemm_wmma_bf16<4><<<N_NODES / 16, 32, 0, stream>>>(
      featb, K0P, K0P, Wb0, F0, hlin0, F0);
  // per-node logits
  node_el_er<<<cdiv((size_t)N_NODES * H0, B), B, 0, stream>>>(
      hlin0, F0, al0, ar0, N_NODES * H0, H0, D0, el0, er0);
  // init accumulators
  fill_u32<<<cdiv((size_t)N_NODES * H0, B), B, 0, stream>>>(emax0, ENC_NEG_INF, (size_t)N_NODES * H0);
  fill_u32<<<cdiv((size_t)N_NODES * H0, B), B, 0, stream>>>((unsigned*)den0, 0u, (size_t)N_NODES * H0);
  fill_u32<<<cdiv((size_t)N_NODES * F0, B), B, 0, stream>>>((unsigned*)hout0, 0u, (size_t)N_NODES * F0);
  // edge softmax + aggregation
  edge_lrelu_max<<<cdiv(N_EDGES, B), B, 0, stream>>>(src, dst, el0, er0, N_EDGES, H0, ebuf0, emax0);
  decode_emax<<<cdiv((size_t)N_NODES * H0, B), B, 0, stream>>>(emax0, (size_t)N_NODES * H0);
  edge_exp_sum<<<cdiv(N_EDGES, B), B, 0, stream>>>(dst, (const float*)emax0, N_EDGES, H0, ebuf0, den0);
  edge_aggregate<<<cdiv((size_t)N_EDGES * H0, B), B, 0, stream>>>(
      src, dst, ebuf0, den0, hlin0, F0, N_EDGES * H0, H0, D0, hout0, F0);
  // activation -> bf16 layer-1 A operand
  elu_to_bf16<<<cdiv((size_t)N_NODES * F0, B), B, 0, stream>>>(hout0, h1b, (size_t)N_NODES * F0);

  // ==================== Layer 1 ====================
  cvt_pad_bf16<<<cdiv((size_t)IN1 * NP1, B), B, 0, stream>>>(
      W1, IN1, D1, NP1, Wb1, (size_t)IN1 * NP1);
  // hlin1 = elu(h0) @ W1   (WMMA, 3125 strips x 1 N-tile; cols 7..15 zero)
  gemm_wmma_bf16<1><<<N_NODES / 16, 32, 0, stream>>>(
      h1b, IN1, IN1, Wb1, NP1, hlin1, NP1);
  node_el_er<<<cdiv((size_t)N_NODES, B), B, 0, stream>>>(
      hlin1, NP1, al1, ar1, N_NODES, H1, D1, el1, er1);
  fill_u32<<<cdiv((size_t)N_NODES, B), B, 0, stream>>>(emax1, ENC_NEG_INF, (size_t)N_NODES);
  fill_u32<<<cdiv((size_t)N_NODES, B), B, 0, stream>>>((unsigned*)den1, 0u, (size_t)N_NODES);
  fill_u32<<<cdiv((size_t)out_size, B), B, 0, stream>>>((unsigned*)out, 0u, (size_t)out_size);
  edge_lrelu_max<<<cdiv(N_EDGES, B), B, 0, stream>>>(src, dst, el1, er1, N_EDGES, H1, ebuf1, emax1);
  decode_emax<<<cdiv((size_t)N_NODES, B), B, 0, stream>>>(emax1, (size_t)N_NODES);
  edge_exp_sum<<<cdiv(N_EDGES, B), B, 0, stream>>>(dst, (const float*)emax1, N_EDGES, H1, ebuf1, den1);
  // H1==1 -> mean over heads is identity: accumulate straight into d_out [N,7]
  edge_aggregate<<<cdiv(N_EDGES, B), B, 0, stream>>>(
      src, dst, ebuf1, den1, hlin1, NP1, N_EDGES, H1, D1, out, D1);
}
